// MTREncoder_31602369364140
// MI455X (gfx1250) — matrix operand order repaired and checked by reading, not compiled, not run on hardware
//
#include <hip/hip_runtime.h>
#include <math.h>

typedef __attribute__((ext_vector_type(16))) _Float16 v16h;
typedef __attribute__((ext_vector_type(8)))  float    v8f;
typedef _Float16 f16;

#define B_     16
#define NO_    128
#define T_     11
#define AA_    29
#define NPOLY_ 768
#define PT_    20
#define AM_    9
#define D_     256
#define H_     8
#define L_     6
#define K_     16
#define NQ_    64
#define N_     896
#define BN_    (B_*N_)        // 14336 tokens
#define BNT_   (B_*NO_*T_)    // 22528 agent point-rows
#define BPP_   (B_*NPOLY_*PT_)// 245760 map point-rows

// ---------------------------------------------------------------------------
// WMMA GEMM: C[M,N] = act(A[M,K](f16) @ Wt[K,N](f16) + bias[N])
// One wave -> 32x64 C tile (2 A-frags x 4 B-frags = 8 WMMA per k-step).
// Requires M%128==0, N%64==0, K%32==0.
// ---------------------------------------------------------------------------
__global__ __launch_bounds__(128) void k_gemm(
    const f16* __restrict__ A, const f16* __restrict__ Wt,
    const float* __restrict__ bias, float* __restrict__ Cf,
    f16* __restrict__ Ch, int M, int N, int K, int relu)
{
  const int lane = threadIdx.x;                      // 0..31
  const int m0 = blockIdx.y * 128 + threadIdx.y * 32;
  const int n0 = blockIdx.x * 64;
  if (m0 >= M) return;

  v8f z = {};
  v8f acc00 = z, acc01 = z, acc02 = z, acc03 = z;
  v8f acc10 = z, acc11 = z, acc12 = z, acc13 = z;

  const int lo = (lane < 16) ? 0 : 8;                // ISA 16x32 f16 A layout
  const f16* Arow0 = A + (long)(m0 + (lane & 15)) * K;
  const f16* Arow1 = Arow0 + (long)16 * K;

  for (int k0 = 0; k0 < K; k0 += 32) {
    v16h a0, a1;
    f16* a0h = (f16*)&a0;
    f16* a1h = (f16*)&a1;
    const f16* ap0 = Arow0 + k0;
    const f16* ap1 = Arow1 + k0;
#pragma unroll
    for (int p = 0; p < 8; ++p) {
      int kk = (p < 4) ? (lo + 2 * p) : (16 + lo + 2 * (p - 4));
      a0h[2 * p]     = ap0[kk];
      a0h[2 * p + 1] = ap0[kk + 1];
      a1h[2 * p]     = ap1[kk];
      a1h[2 * p + 1] = ap1[kk + 1];
    }
    // B 32x16 frags: lane = N (lane&15), 16 K elements in same order as A
    const f16* Wcol = Wt + (long)k0 * N + n0 + (lane & 15);
    v16h b0, b1, b2, b3;
#pragma unroll
    for (int s = 0; s < 16; ++s) {
      int kk = (s < 8) ? (lo + s) : (16 + lo + (s - 8));
      const f16* wp = Wcol + (long)kk * N;
      b0[s] = wp[0];
      b1[s] = wp[16];
      b2[s] = wp[32];
      b3[s] = wp[48];
    }
    acc00 = __builtin_amdgcn_wmma_f32_16x16x32_f16(false, a0, false, b0, (short)0, acc00, false, false);
    acc01 = __builtin_amdgcn_wmma_f32_16x16x32_f16(false, a0, false, b1, (short)0, acc01, false, false);
    acc02 = __builtin_amdgcn_wmma_f32_16x16x32_f16(false, a0, false, b2, (short)0, acc02, false, false);
    acc03 = __builtin_amdgcn_wmma_f32_16x16x32_f16(false, a0, false, b3, (short)0, acc03, false, false);
    acc10 = __builtin_amdgcn_wmma_f32_16x16x32_f16(false, a1, false, b0, (short)0, acc10, false, false);
    acc11 = __builtin_amdgcn_wmma_f32_16x16x32_f16(false, a1, false, b1, (short)0, acc11, false, false);
    acc12 = __builtin_amdgcn_wmma_f32_16x16x32_f16(false, a1, false, b2, (short)0, acc12, false, false);
    acc13 = __builtin_amdgcn_wmma_f32_16x16x32_f16(false, a1, false, b3, (short)0, acc13, false, false);
  }

  // Epilogue: C layout VGPR r -> M = r + (lane<16?0:8), N = lane&15
  v8f accs[2][4] = {{acc00, acc01, acc02, acc03}, {acc10, acc11, acc12, acc13}};
  const int cln = lane & 15;
  const int cmo = (lane < 16) ? 0 : 8;
#pragma unroll
  for (int i = 0; i < 2; ++i) {
#pragma unroll
    for (int j = 0; j < 4; ++j) {
      const int cn = n0 + 16 * j + cln;
      const float bv = bias ? bias[cn] : 0.f;
      const int cmb = m0 + 16 * i + cmo;
#pragma unroll
      for (int r = 0; r < 8; ++r) {
        float v = accs[i][j][r] + bv;
        if (relu) v = fmaxf(v, 0.f);
        long ci = (long)(cmb + r) * N + cn;
        if (Cf) Cf[ci] = v;
        if (Ch) Ch[ci] = (f16)v;
      }
    }
  }
}

// Wt[k*Nw + n] = (k < Kw) ? W[n*Kw + k] : 0   (f32 -> f16, k-major, zero-pad K)
__global__ void k_convert_wt(const float* __restrict__ W, f16* __restrict__ Wt,
                             int Nw, int Kw, int Kp)
{
  long idx = (long)blockIdx.x * blockDim.x + threadIdx.x;
  if (idx >= (long)Kp * Nw) return;
  int k = idx / Nw, n = idx % Nw;
  Wt[idx] = (k < Kw) ? (f16)W[(long)n * Kw + k] : (f16)0.f;
}

// Agent input rows: [ref_point(2), ref_dist(1), traj[3:29](26), mask(1), pad(2)]
__global__ void k_obj_in(const float* __restrict__ obj, const float* __restrict__ ip,
                         const int* __restrict__ refidx,
                         const unsigned char* __restrict__ mask, f16* __restrict__ out)
{
  int r = blockIdx.x * blockDim.x + threadIdx.x;
  if (r >= BNT_) return;
  const float* tr = obj + (long)r * AA_;
  int q = refidx[r];
  float ipx = ip[q * 2], ipy = ip[q * 2 + 1];
  float dx = tr[0] - ipx, dy = tr[1] - ipy;
  f16* o = out + (long)r * 32;
  o[0] = (f16)ipx; o[1] = (f16)ipy; o[2] = (f16)sqrtf(dx * dx + dy * dy);
#pragma unroll
  for (int i = 0; i < 26; ++i) o[3 + i] = (f16)tr[3 + i];
  o[29] = (f16)(mask[r] ? 1.f : 0.f);
  o[30] = (f16)0.f; o[31] = (f16)0.f;
}

__global__ void k_map_in(const float* __restrict__ mp, f16* __restrict__ out)
{
  int r = blockIdx.x * blockDim.x + threadIdx.x;
  if (r >= BPP_) return;
  const float* s = mp + (long)r * AM_;
  f16* o = out + (long)r * 32;
#pragma unroll
  for (int i = 0; i < 9; ++i) o[i] = (f16)s[i];
#pragma unroll
  for (int i = 9; i < 32; ++i) o[i] = (f16)0.f;
}

// masked write + masked max-pool broadcast concat: out[.,0:C]=mask?f:0, out[.,C:2C]=pool
__global__ void k_pool_concat(const f16* __restrict__ f, const unsigned char* __restrict__ mask,
                              f16* __restrict__ out, int C, int Tn)
{
  int g = blockIdx.x, c = threadIdx.x;
  float m = -3.0e38f;
  for (int t = 0; t < Tn; ++t) {
    long r = (long)g * Tn + t;
    float v = mask[r] ? (float)f[r * C + c] : 0.f;
    out[r * (2 * C) + c] = (f16)v;
    m = fmaxf(m, v);
  }
  f16 mh = (f16)m;
  for (int t = 0; t < Tn; ++t) out[((long)g * Tn + t) * (2 * C) + C + c] = mh;
}

__global__ void k_pool(const f16* __restrict__ f, const unsigned char* __restrict__ mask,
                       f16* __restrict__ out, int C, int Tn)
{
  int g = blockIdx.x, c = threadIdx.x;
  float m = -3.0e38f;
  for (int t = 0; t < Tn; ++t) {
    long r = (long)g * Tn + t;
    float v = mask[r] ? (float)f[r * C + c] : 0.f;
    m = fmaxf(m, v);
  }
  out[(long)g * C + c] = (f16)m;
}

// scatter pointnet output rows into token stream x, zeroing rows with no valid points
__global__ void k_scatter(const float* __restrict__ tmp, const unsigned char* __restrict__ mask,
                          float* __restrict__ x, int rowsPerB, int tokOff, int Tn, int rows)
{
  long idx = (long)blockIdx.x * blockDim.x + threadIdx.x;
  if (idx >= (long)rows * D_) return;
  int row = idx / D_, c = idx % D_;
  int b = row / rowsPerB, e = row % rowsPerB;
  bool any = false;
  for (int t = 0; t < Tn; ++t) any = any || mask[(long)row * Tn + t];
  x[((long)b * N_ + tokOff + e) * D_ + c] = any ? tmp[idx] : 0.f;
}

__global__ void k_pos(const float* __restrict__ lastpos, const float* __restrict__ center,
                      float* __restrict__ pos)
{
  long idx = (long)blockIdx.x * blockDim.x + threadIdx.x;
  if (idx >= (long)BN_ * 3) return;
  int t = idx / 3, d = idx % 3;
  int b = t / N_, n = t % N_;
  pos[idx] = (n < NO_) ? lastpos[((long)b * NO_ + n) * 3 + d]
                       : center[((long)b * NPOLY_ + (n - NO_)) * 3 + d];
}

__global__ void k_knn(const float* __restrict__ pos, int* __restrict__ nbr)
{
  int t = blockIdx.x * blockDim.x + threadIdx.x;
  if (t >= BN_) return;
  int b = t / N_;
  float qx = pos[(long)t * 3], qy = pos[(long)t * 3 + 1], qz = pos[(long)t * 3 + 2];
  float bd[K_]; int bi[K_];
#pragma unroll
  for (int i = 0; i < K_; ++i) { bd[i] = 3.4e38f; bi[i] = 0; }
  long base = (long)b * N_;
  for (int j = 0; j < N_; ++j) {
    const float* p = pos + (base + j) * 3;
    float dx = qx - p[0], dy = qy - p[1], dz = qz - p[2];
    float d2 = dx * dx + dy * dy + dz * dz;
    if (d2 < bd[K_ - 1]) {
      int pz = K_ - 1;
      while (pz > 0 && bd[pz - 1] > d2) { bd[pz] = bd[pz - 1]; bi[pz] = bi[pz - 1]; --pz; }
      bd[pz] = d2; bi[pz] = j;
    }
  }
  for (int k = 0; k < K_; ++k) nbr[(long)t * K_ + k] = bi[k];
}

__global__ void k_pe(const float* __restrict__ pos, float* __restrict__ pe)
{
  long idx = (long)blockIdx.x * blockDim.x + threadIdx.x;
  if (idx >= (long)BN_ * D_) return;
  int t = idx / D_, j = idx % D_;
  float v = (j < 128) ? pos[(long)t * 3 + 1] : pos[(long)t * 3];  // y first, then x
  int jj = j & 127;
  float dim_t = powf(10000.f, (2.0f * (float)(jj >> 1)) / 128.0f);
  float p = v * 6.283185307179586f / dim_t;
  pe[idx] = (jj & 1) ? cosf(p) : sinf(p);
}

__global__ void k_prep(const float* __restrict__ x, const float* __restrict__ pe,
                       f16* __restrict__ srcpe, f16* __restrict__ xf)
{
  long idx = (long)blockIdx.x * blockDim.x + threadIdx.x;
  if (idx >= (long)BN_ * D_) return;
  float xv = x[idx];
  srcpe[idx] = (f16)(xv + pe[idx]);
  xf[idx]    = (f16)xv;
}

// local attention: one thread per (token, head); qk buf: [BN, 512] = [q | k]
__global__ void k_attn(const f16* __restrict__ qk, const f16* __restrict__ vbuf,
                       const int* __restrict__ nbr, f16* __restrict__ o)
{
  int gid = blockIdx.x * blockDim.x + threadIdx.x;
  if (gid >= BN_ * H_) return;
  int t = gid / H_, h = gid % H_;
  int b = t / N_;
  const f16* q = qk + (long)t * (2 * D_) + h * 32;
  float qv[32];
#pragma unroll
  for (int d = 0; d < 32; ++d) qv[d] = (float)q[d];
  int nbs[K_];
  for (int kk = 0; kk < K_; ++kk) nbs[kk] = b * N_ + nbr[(long)t * K_ + kk];
  float lg[K_], mx = -3.0e38f;
  for (int kk = 0; kk < K_; ++kk) {
    const f16* kp = qk + (long)nbs[kk] * (2 * D_) + D_ + h * 32;
    float s = 0.f;
#pragma unroll
    for (int d = 0; d < 32; ++d) s += qv[d] * (float)kp[d];
    s *= 0.17677669529663687f;  // 1/sqrt(hd=32)
    lg[kk] = s; mx = fmaxf(mx, s);
  }
  float sum = 0.f;
  for (int kk = 0; kk < K_; ++kk) { float e = __expf(lg[kk] - mx); lg[kk] = e; sum += e; }
  float inv = 1.f / sum;
#pragma unroll
  for (int d = 0; d < 32; ++d) {
    float acc = 0.f;
    for (int kk = 0; kk < K_; ++kk)
      acc += lg[kk] * (float)vbuf[(long)nbs[kk] * D_ + h * 32 + d];
    o[(long)t * D_ + h * 32 + d] = (f16)(acc * inv);
  }
}

// x = LN(x + add); optionally emit f16 copy for next GEMM
__global__ __launch_bounds__(256) void k_ln(float* __restrict__ x, const float* __restrict__ add,
                                            const float* __restrict__ g, const float* __restrict__ b,
                                            f16* __restrict__ xh)
{
  __shared__ float red[256];
  long base = (long)blockIdx.x * D_;
  int c = threadIdx.x;
  float v = x[base + c] + add[base + c];
  red[c] = v; __syncthreads();
  for (int s = 128; s > 0; s >>= 1) { if (c < s) red[c] += red[c + s]; __syncthreads(); }
  float mean = red[0] * (1.0f / D_);
  __syncthreads();
  float d = v - mean;
  red[c] = d * d; __syncthreads();
  for (int s = 128; s > 0; s >>= 1) { if (c < s) red[c] += red[c + s]; __syncthreads(); }
  float y = d * rsqrtf(red[0] * (1.0f / D_) + 1e-5f) * g[c] + b[c];
  x[base + c] = y;
  if (xh) xh[base + c] = (f16)y;
}

// ---------------------------------------------------------------------------
extern "C" void kernel_launch(void* const* d_in, const int* in_sizes, int n_in,
                              void* d_out, int out_size, void* d_ws, size_t ws_size,
                              hipStream_t stream)
{
  (void)in_sizes; (void)n_in; (void)out_size; (void)ws_size;
  const float* obj_trajs  = (const float*)d_in[0];
  const float* obj_last   = (const float*)d_in[1];
  const float* map_poly   = (const float*)d_in[2];
  const float* map_center = (const float*)d_in[3];
  const float* ipts       = (const float*)d_in[4];
  const float* a_pre_w  = (const float*)d_in[5];  const float* a_pre_b  = (const float*)d_in[6];
  const float* a_mid1_w = (const float*)d_in[7];  const float* a_mid1_b = (const float*)d_in[8];
  const float* a_mid2_w = (const float*)d_in[9];  const float* a_mid2_b = (const float*)d_in[10];
  const float* a_out1_w = (const float*)d_in[11]; const float* a_out1_b = (const float*)d_in[12];
  const float* a_out2_w = (const float*)d_in[13]; const float* a_out2_b = (const float*)d_in[14];
  const float* m_pre1_w = (const float*)d_in[15]; const float* m_pre1_b = (const float*)d_in[16];
  const float* m_pre2_w = (const float*)d_in[17]; const float* m_pre2_b = (const float*)d_in[18];
  const float* m_pre3_w = (const float*)d_in[19]; const float* m_pre3_b = (const float*)d_in[20];
  const float* m_mid1_w = (const float*)d_in[21]; const float* m_mid1_b = (const float*)d_in[22];
  const float* m_mid2_w = (const float*)d_in[23]; const float* m_mid2_b = (const float*)d_in[24];
  const float* m_out1_w = (const float*)d_in[25]; const float* m_out1_b = (const float*)d_in[26];
  const float* m_out2_w = (const float*)d_in[27]; const float* m_out2_b = (const float*)d_in[28];
  const float* qkv_w  = (const float*)d_in[29];   const float* qkv_b  = (const float*)d_in[30];
  const float* out_w  = (const float*)d_in[31];   const float* out_b  = (const float*)d_in[32];
  const float* ffn1_w = (const float*)d_in[33];   const float* ffn1_b = (const float*)d_in[34];
  const float* ffn2_w = (const float*)d_in[35];   const float* ffn2_b = (const float*)d_in[36];
  const float* ln1_g = (const float*)d_in[37];    const float* ln1_b = (const float*)d_in[38];
  const float* ln2_g = (const float*)d_in[39];    const float* ln2_b = (const float*)d_in[40];
  const int* refidx = (const int*)d_in[41];
  const unsigned char* omask = (const unsigned char*)d_in[42];
  const unsigned char* mmask = (const unsigned char*)d_in[43];
  float* x = (float*)d_out;  // [B, N, D] residual stream lives in d_out

  char* wsB = (char*)d_ws;
  size_t off = 0;
  auto alloc = [&](size_t bytes) -> char* {
    char* p = wsB + off;
    off = (off + bytes + 255) & ~(size_t)255;
    return p;
  };

  // -------- f16 k-major weights --------
  f16* wApre = (f16*)alloc((size_t)32 * 256 * 2);
  f16* wAm1  = (f16*)alloc((size_t)512 * 256 * 2);
  f16* wAm2  = (f16*)alloc((size_t)256 * 256 * 2);
  f16* wAo1  = (f16*)alloc((size_t)256 * 256 * 2);
  f16* wAo2  = (f16*)alloc((size_t)256 * 256 * 2);
  f16* wMp1  = (f16*)alloc((size_t)32 * 64 * 2);
  f16* wMp2  = (f16*)alloc((size_t)64 * 64 * 2);
  f16* wMp3  = (f16*)alloc((size_t)64 * 64 * 2);
  f16* wMm1  = (f16*)alloc((size_t)128 * 64 * 2);
  f16* wMm2  = (f16*)alloc((size_t)64 * 64 * 2);
  f16* wMo1  = (f16*)alloc((size_t)64 * 64 * 2);
  f16* wMo2  = (f16*)alloc((size_t)64 * 256 * 2);
  f16 *wQK[L_], *wV[L_], *wO[L_], *wF1[L_], *wF2[L_];
  for (int l = 0; l < L_; ++l) {
    wQK[l] = (f16*)alloc((size_t)256 * 512 * 2);
    wV[l]  = (f16*)alloc((size_t)256 * 256 * 2);
    wO[l]  = (f16*)alloc((size_t)256 * 256 * 2);
    wF1[l] = (f16*)alloc((size_t)256 * 1024 * 2);
    wF2[l] = (f16*)alloc((size_t)1024 * 256 * 2);
  }

  // -------- persistent + scratch activation buffers --------
  float* PE  = (float*)alloc((size_t)BN_ * D_ * 4);
  float* POS = (float*)alloc((size_t)BN_ * 3 * 4);
  int*   NBR = (int*)alloc((size_t)BN_ * K_ * 4);
  float* TMP = (float*)alloc((size_t)12288 * 256 * 4);      // pointnet out2 staging
  f16*   S0  = (f16*)alloc((size_t)12288 * 64 * 2 * 2);     // pooled (max of both nets)
  f16*   S1  = (f16*)alloc((size_t)12288 * 64 * 2 * 2);     // out1 result
  char*  P0  = alloc((size_t)32 << 20);
  char*  P1  = alloc((size_t)64 << 20);
  char*  P2  = alloc((size_t)32 << 20);

  auto conv = [&](const float* W, f16* Wt, int Nw, int Kw, int Kp) {
    long n = (long)Kp * Nw;
    k_convert_wt<<<dim3((unsigned)((n + 255) / 256)), dim3(256), 0, stream>>>(W, Wt, Nw, Kw, Kp);
  };
  auto gemm = [&](const f16* A, const f16* Wt, const float* bias, float* Cf, f16* Ch,
                  int M, int N, int K, int relu) {
    k_gemm<<<dim3(N / 64, M / 128), dim3(32, 4), 0, stream>>>(A, Wt, bias, Cf, Ch, M, N, K, relu);
  };

  // -------- weight conversion (every call; deterministic) --------
  conv(a_pre_w,  wApre, 256, 30, 32);
  conv(a_mid1_w, wAm1, 256, 512, 512);
  conv(a_mid2_w, wAm2, 256, 256, 256);
  conv(a_out1_w, wAo1, 256, 256, 256);
  conv(a_out2_w, wAo2, 256, 256, 256);
  conv(m_pre1_w, wMp1, 64, 9, 32);
  conv(m_pre2_w, wMp2, 64, 64, 64);
  conv(m_pre3_w, wMp3, 64, 64, 64);
  conv(m_mid1_w, wMm1, 64, 128, 128);
  conv(m_mid2_w, wMm2, 64, 64, 64);
  conv(m_out1_w, wMo1, 64, 64, 64);
  conv(m_out2_w, wMo2, 256, 64, 64);
  for (int l = 0; l < L_; ++l) {
    conv(qkv_w + (size_t)l * 768 * 256,             wQK[l], 512, 256, 256);
    conv(qkv_w + (size_t)l * 768 * 256 + 512 * 256, wV[l],  256, 256, 256);
    conv(out_w + (size_t)l * 256 * 256,             wO[l],  256, 256, 256);
    conv(ffn1_w + (size_t)l * 1024 * 256,           wF1[l], 1024, 256, 256);
    conv(ffn2_w + (size_t)l * 256 * 1024,           wF2[l], 256, 1024, 1024);
  }

  // -------- agent PointNet --------
  f16* objin = (f16*)P0;                              // [22528, 32]
  k_obj_in<<<(BNT_ + 255) / 256, 256, 0, stream>>>(obj_trajs, ipts, refidx, omask, objin);
  f16* f1 = (f16*)P1;                                 // [22528, 256]
  gemm(objin, wApre, a_pre_b, nullptr, f1, BNT_, 256, 32, 1);
  f16* amid = (f16*)P2;                               // [22528, 512]
  k_pool_concat<<<B_ * NO_, 256, 0, stream>>>(f1, omask, amid, 256, T_);
  f16* g2 = (f16*)P0;                                 // [22528, 256]
  gemm(amid, wAm1, a_mid1_b, nullptr, g2, BNT_, 256, 512, 1);
  f16* g3 = (f16*)P1;                                 // [22528, 256]
  gemm(g2, wAm2, a_mid2_b, nullptr, g3, BNT_, 256, 256, 1);
  k_pool<<<B_ * NO_, 256, 0, stream>>>(g3, omask, S0, 256, T_);
  gemm(S0, wAo1, a_out1_b, nullptr, S1, B_ * NO_, 256, 256, 1);
  gemm(S1, wAo2, a_out2_b, TMP, nullptr, B_ * NO_, 256, 256, 0);
  k_scatter<<<(B_ * NO_ * D_ + 255) / 256, 256, 0, stream>>>(TMP, omask, x, NO_, 0, T_, B_ * NO_);

  // -------- map PointNet --------
  f16* mapin = (f16*)P0;                              // [245760, 32]
  k_map_in<<<(BPP_ + 255) / 256, 256, 0, stream>>>(map_poly, mapin);
  f16* m1 = (f16*)P1;
  gemm(mapin, wMp1, m_pre1_b, nullptr, m1, BPP_, 64, 32, 1);
  f16* m2 = (f16*)P2;
  gemm(m1, wMp2, m_pre2_b, nullptr, m2, BPP_, 64, 64, 1);
  f16* m3 = (f16*)P0;
  gemm(m2, wMp3, m_pre3_b, nullptr, m3, BPP_, 64, 64, 1);
  f16* mmid = (f16*)P1;                               // [245760, 128]
  k_pool_concat<<<B_ * NPOLY_, 64, 0, stream>>>(m3, mmask, mmid, 64, PT_);
  f16* mm1 = (f16*)P2;
  gemm(mmid, wMm1, m_mid1_b, nullptr, mm1, BPP_, 64, 128, 1);
  f16* mm2 = (f16*)P0;
  gemm(mm1, wMm2, m_mid2_b, nullptr, mm2, BPP_, 64, 64, 1);
  k_pool<<<B_ * NPOLY_, 64, 0, stream>>>(mm2, mmask, S0, 64, PT_);
  gemm(S0, wMo1, m_out1_b, nullptr, S1, B_ * NPOLY_, 64, 64, 1);
  gemm(S1, wMo2, m_out2_b, TMP, nullptr, B_ * NPOLY_, 256, 64, 0);
  k_scatter<<<(B_ * NPOLY_ * D_ + 255) / 256, 256, 0, stream>>>(TMP, mmask, x, NPOLY_, NO_, PT_, B_ * NPOLY_);

  // -------- positions, KNN, sine PE --------
  k_pos<<<(BN_ * 3 + 255) / 256, 256, 0, stream>>>(obj_last, map_center, POS);
  k_knn<<<(BN_ + 127) / 128, 128, 0, stream>>>(POS, NBR);
  k_pe<<<(BN_ * D_ + 255) / 256, 256, 0, stream>>>(POS, PE);

  // -------- transformer layers --------
  f16* srcpe = (f16*)P0;                               // [BN, 256]
  f16* xf    = (f16*)(P0 + ((size_t)8 << 20));         // [BN, 256]
  f16* qkb   = (f16*)P1;                               // [BN, 512]
  f16* vb    = (f16*)(P1 + ((size_t)16 << 20));        // [BN, 256]
  f16* ob    = (f16*)P2;                               // [BN, 256]
  float* prj = (float*)(P2 + ((size_t)8 << 20));       // [BN, 256] f32
  f16* hb    = (f16*)P1;                               // [BN, 1024] (reuses qk/v)

  for (int l = 0; l < L_; ++l) {
    k_prep<<<(BN_ * D_ + 255) / 256, 256, 0, stream>>>(x, PE, srcpe, xf);
    gemm(srcpe, wQK[l], qkv_b + (size_t)l * 768,       nullptr, qkb, BN_, 512, 256, 0);
    gemm(xf,    wV[l],  qkv_b + (size_t)l * 768 + 512, nullptr, vb,  BN_, 256, 256, 0);
    k_attn<<<(BN_ * H_ + 255) / 256, 256, 0, stream>>>(qkb, vb, NBR, ob);
    gemm(ob, wO[l], out_b + (size_t)l * 256, prj, nullptr, BN_, 256, 256, 0);
    k_ln<<<BN_, 256, 0, stream>>>(x, prj, ln1_g + (size_t)l * 256, ln1_b + (size_t)l * 256, xf);
    gemm(xf, wF1[l], ffn1_b + (size_t)l * 1024, nullptr, hb, BN_, 1024, 256, 1);
    gemm(hb, wF2[l], ffn2_b + (size_t)l * 256, prj, nullptr, BN_, 256, 1024, 0);
    k_ln<<<BN_, 256, 0, stream>>>(x, prj, ln2_g + (size_t)l * 256, ln2_b + (size_t)l * 256, nullptr);
  }
}
